// Reg3dLoss_65146063946227
// MI455X (gfx1250) — compile-verified, matrix operand order checked by reading, not run on hardware
//
#include <hip/hip_runtime.h>

#define PI_F 3.1415926f

typedef __attribute__((ext_vector_type(2)))  float    v2f;
typedef __attribute__((ext_vector_type(8)))  float    v8f;
typedef __attribute__((ext_vector_type(16))) _Float16 v16h;

__device__ __forceinline__ void inv3(const float a[9], float r[9]) {
    float c00 =   a[4]*a[8] - a[5]*a[7];
    float c01 = -(a[3]*a[8] - a[5]*a[6]);
    float c02 =   a[3]*a[7] - a[4]*a[6];
    float det = a[0]*c00 + a[1]*c01 + a[2]*c02;
    float id  = 1.0f / det;
    r[0] =  c00 * id;
    r[1] = -(a[1]*a[8] - a[2]*a[7]) * id;
    r[2] =  (a[1]*a[5] - a[2]*a[4]) * id;
    r[3] =  c01 * id;
    r[4] =  (a[0]*a[8] - a[2]*a[6]) * id;
    r[5] = -(a[0]*a[5] - a[2]*a[3]) * id;
    r[6] =  c02 * id;
    r[7] = -(a[0]*a[7] - a[1]*a[6]) * id;
    r[8] =  (a[0]*a[4] - a[1]*a[3]) * id;
}

__device__ __forceinline__ float obj_loss(
    int obj, int Kobj, int C, int HW,
    const float* __restrict__ output,
    const int*   __restrict__ ind,
    const int*   __restrict__ cls_id,
    const int*   __restrict__ reg_mask,
    const int*   __restrict__ flip_mask,
    const float* __restrict__ ct3d,
    const float* __restrict__ reg_3d,
    const float* __restrict__ roty,
    const float* __restrict__ Kmat,
    const float* __restrict__ trans)
{
    const float XS[8] = {1.f,1.f,-1.f,-1.f,1.f,1.f,-1.f,-1.f};
    const float YS[8] = {0.f,0.f,0.f,0.f,1.f,1.f,1.f,1.f};
    const float ZS[8] = {1.f,-1.f,-1.f,1.f,1.f,-1.f,-1.f,1.f};
    const float DIMS_AVG[9] = {1.63f,1.53f,3.88f, 1.73f,0.67f,0.88f, 1.70f,0.58f,1.78f};

    int   b   = obj / Kobj;
    float m   = (float)reg_mask[obj];
    float fm  = (float)flip_mask[obj];
    int   cls = cls_id[obj];
    long  idx = (long)ind[obj];

    // Gather pred channels (strided by HW) and masked targets.
    float pred[8], tgt[8];
    long base = (long)b * (long)C * (long)HW + idx;
    #pragma unroll
    for (int c = 0; c < 8; ++c) pred[c] = output[base + (long)c * (long)HW] * m;
    #pragma unroll
    for (int c = 0; c < 8; ++c) tgt[c]  = reg_3d[(long)obj * 8 + c] * m;

    // Per-batch 3x3 inverses (adjugate): [[trans],[0,0,1]]^-1 and K[:, :3]^-1.
    const float* T = trans + (long)b * 6;
    float Tm[9] = {T[0],T[1],T[2], T[3],T[4],T[5], 0.f,0.f,1.f};
    float Tinv[9]; inv3(Tm, Tinv);
    const float* Kb = Kmat + (long)b * 12;
    float Km[9] = {Kb[0],Kb[1],Kb[2], Kb[4],Kb[5],Kb[6], Kb[8],Kb[9],Kb[10]};
    float Kinv[9]; inv3(Km, Kinv);

    float ctx = ct3d[(long)obj*2+0], cty = ct3d[(long)obj*2+1];
    float pd = pred[0]*16.32f + 28.01f;
    float gd = tgt[0] *16.32f + 28.01f;

    float p_loc[3], g_loc[3];
    {
        float hx = ctx + pred[1], hy = cty + pred[2];
        float ix = (Tinv[0]*hx + Tinv[1]*hy + Tinv[2]) * pd;
        float iy = (Tinv[3]*hx + Tinv[4]*hy + Tinv[5]) * pd;
        float iz = (Tinv[6]*hx + Tinv[7]*hy + Tinv[8]) * pd;
        p_loc[0] = Kinv[0]*ix + Kinv[1]*iy + Kinv[2]*iz;
        p_loc[1] = Kinv[3]*ix + Kinv[4]*iy + Kinv[5]*iz;
        p_loc[2] = Kinv[6]*ix + Kinv[7]*iy + Kinv[8]*iz;
    }
    {
        float hx = ctx + tgt[1], hy = cty + tgt[2];
        float ix = (Tinv[0]*hx + Tinv[1]*hy + Tinv[2]) * gd;
        float iy = (Tinv[3]*hx + Tinv[4]*hy + Tinv[5]) * gd;
        float iz = (Tinv[6]*hx + Tinv[7]*hy + Tinv[8]) * gd;
        g_loc[0] = Kinv[0]*ix + Kinv[1]*iy + Kinv[2]*iz;
        g_loc[1] = Kinv[3]*ix + Kinv[4]*iy + Kinv[5]*iz;
        g_loc[2] = Kinv[6]*ix + Kinv[7]*iy + Kinv[8]*iz;
    }

    float p_dim[3], g_dim[3];
    #pragma unroll
    for (int i = 0; i < 3; ++i) {
        p_dim[i] = expf(pred[3+i]) * DIMS_AVG[cls*3+i];
        g_dim[i] = expf(tgt[3+i])  * DIMS_AVG[cls*3+i];
    }
    p_loc[1] += 0.5f * p_dim[0];
    g_loc[1] += 0.5f * g_dim[0];

    // Orientation decode (ori_dec) using g_loc, flip mask.
    float off   = atanf(g_loc[0] / g_loc[2] + 1e-7f);
    float alpha = atanf(pred[6] / (pred[7] + 1e-7f));
    alpha += (pred[7] >= 0.f) ? -0.5f*PI_F : 0.5f*PI_F;
    float r = alpha + off;
    if      (r >  PI_F) r -= 2.f*PI_F;
    else if (r < -PI_F) r += 2.f*PI_F;
    float rf = fm * r;
    if      (rf > 0.f) rf -= PI_F;
    else if (rf < 0.f) rf += PI_F;
    float p_r = m * (fm*rf + (1.f - fm)*r);
    float g_r = m * roty[obj];

    float cg = cosf(g_r), sg = sinf(g_r);
    float cp = cosf(p_r), sp = sinf(p_r);

    // Three corner-box L1 terms vs gt_box, fused per corner.
    float L = 0.f;
    #pragma unroll
    for (int i = 0; i < 8; ++i) {
        float gx = 0.5f*g_dim[2]*XS[i];
        float gy = -g_dim[1]*YS[i];
        float gz = 0.5f*g_dim[0]*ZS[i];
        float gtx =  cg*gx + sg*gz + g_loc[0];
        float gty =  gy            + g_loc[1];
        float gtz = -sg*gx + cg*gz + g_loc[2];
        // term1: pred rotation, gt dims/loc
        L += fabsf( cp*gx + sp*gz + g_loc[0] - gtx);
        L += fabsf( gy            + g_loc[1] - gty);
        L += fabsf(-sp*gx + cp*gz + g_loc[2] - gtz);
        // term2: pred dims, gt rot/loc
        float px = 0.5f*p_dim[2]*XS[i];
        float py = -p_dim[1]*YS[i];
        float pz = 0.5f*p_dim[0]*ZS[i];
        L += fabsf( cg*px + sg*pz + g_loc[0] - gtx);
        L += fabsf( py            + g_loc[1] - gty);
        L += fabsf(-sg*px + cg*pz + g_loc[2] - gtz);
        // term3: pred loc, gt rot/dims
        L += fabsf( cg*gx + sg*gz + p_loc[0] - gtx);
        L += fabsf( gy            + p_loc[1] - gty);
        L += fabsf(-sg*gx + cg*gz + p_loc[2] - gtz);
    }
    return L;
}

__global__ void zero_out_kernel(float* out) { out[0] = 0.f; }

__global__ void __launch_bounds__(256) decode_loss_kernel(
    const float* __restrict__ output,
    const int*   __restrict__ ind,
    const int*   __restrict__ cls_id,
    const int*   __restrict__ reg_mask,
    const int*   __restrict__ flip_mask,
    const float* __restrict__ ct3d,
    const float* __restrict__ reg_3d,
    const float* __restrict__ roty,
    const float* __restrict__ Kmat,
    const float* __restrict__ trans,
    int N, int Kobj, int C, int HW, float invDenom,
    float* __restrict__ out)
{
    int t  = blockIdx.x * blockDim.x + threadIdx.x;
    int o0 = 2*t, o1 = 2*t + 1;

    float L0 = 0.f, L1 = 0.f;
    if (o0 < N) L0 = obj_loss(o0, Kobj, C, HW, output, ind, cls_id, reg_mask,
                              flip_mask, ct3d, reg_3d, roty, Kmat, trans);
    if (o1 < N) L1 = obj_loss(o1, Kobj, C, HW, output, ind, cls_id, reg_mask,
                              flip_mask, ct3d, reg_3d, roty, Kmat, trans);

    // Wave-level sum of all 64 per-lane partials via one WMMA:
    // A(16x4) holds the 64 values (2 per lane), B = all-ones => D rows = row-sums.
    v8f acc = {};
#if __has_builtin(__builtin_amdgcn_wmma_f32_16x16x4_f32)
    v2f a; a[0] = L0; a[1] = L1;
    v2f bones; bones[0] = 1.0f; bones[1] = 1.0f;
    acc = __builtin_amdgcn_wmma_f32_16x16x4_f32(
        /*neg_a=*/false, a, /*neg_b=*/false, bones,
        /*c_mod=*/(short)0, acc, /*reuse_a=*/false, /*reuse_b=*/false);
#else
    // Fallback: f16 ones-trick (codegen-confirmed builtin); partials fit f16 range.
    v16h a = {};
    a[0] = (_Float16)L0; a[1] = (_Float16)L1;
    v16h bones;
    #pragma unroll
    for (int i = 0; i < 16; ++i) bones[i] = (_Float16)1.0f;
    acc = __builtin_amdgcn_wmma_f32_16x16x32_f16(
        false, a, false, bones, (short)0, acc, false, false);
#endif
    // Each lane's 8 accumulator regs hold rows M=0..7 (lanes 0-15) or 8..15 (lanes 16-31)
    // of the same column; lane n + lane n^16 covers all 16 rows => wave total.
    float s = acc[0]+acc[1]+acc[2]+acc[3]+acc[4]+acc[5]+acc[6]+acc[7];
    s += __shfl_xor(s, 16, 32);

    if ((threadIdx.x & 31) == 0)
        atomicAdd(out, s * invDenom);
}

extern "C" void kernel_launch(void* const* d_in, const int* in_sizes, int n_in,
                              void* d_out, int out_size, void* d_ws, size_t ws_size,
                              hipStream_t stream) {
    const float* output    = (const float*)d_in[0];
    const int*   ind       = (const int*)  d_in[1];
    const int*   cls_id    = (const int*)  d_in[2];
    const int*   reg_mask  = (const int*)  d_in[3];
    const int*   flip_mask = (const int*)  d_in[4];
    const float* ct3d      = (const float*)d_in[5];
    const float* reg_3d    = (const float*)d_in[6];
    const float* roty      = (const float*)d_in[7];
    const float* Kmat      = (const float*)d_in[8];
    const float* trans     = (const float*)d_in[9];
    float* out = (float*)d_out;

    int B    = in_sizes[8] / 12;          // K is (B,3,4)
    int N    = in_sizes[1];               // B*Kobj
    int Kobj = N / B;
    int C    = in_sizes[6] / N;           // reg_3d is (B,Kobj,8) -> 8
    int HW   = in_sizes[0] / (B * C);
    float invDenom = 1.0f / ((float)Kobj * 10.0f);

    zero_out_kernel<<<1, 1, 0, stream>>>(out);

    int nThreads = (N + 1) / 2;           // 2 objects per thread
    int block = 256;
    int grid  = (nThreads + block - 1) / block;
    decode_loss_kernel<<<grid, block, 0, stream>>>(
        output, ind, cls_id, reg_mask, flip_mask, ct3d, reg_3d, roty,
        Kmat, trans, N, Kobj, C, HW, invDenom, out);
}
